// GATv2Layer_44813688766819
// MI455X (gfx1250) — matrix operand order repaired and checked by reading, not compiled, not run on hardware
//
#include <hip/hip_runtime.h>
#include <hip/hip_bf16.h>

typedef __attribute__((ext_vector_type(16))) __bf16 v16bf;
typedef __attribute__((ext_vector_type(8)))  __bf16 v8bf;
typedef __attribute__((ext_vector_type(8)))  float  v8f;
typedef __attribute__((ext_vector_type(4)))  float  v4f;

#define HC 256          // HEADS * OUT_C
#define INC 64
#define NEG_SLOPE 0.2f
#define LN_EPS 1e-5f

// ---------- helpers: order-preserving float<->uint encoding for atomic max ----------
__device__ __forceinline__ unsigned enc_f32(float f) {
    unsigned b = __float_as_uint(f);
    return (b & 0x80000000u) ? ~b : (b | 0x80000000u);
}
__device__ __forceinline__ float dec_f32(unsigned k) {
    unsigned b = (k & 0x80000000u) ? (k ^ 0x80000000u) : ~k;
    return __uint_as_float(b);
}

// ---------- 1) convert x to bf16 ----------
__global__ void cvt_x_bf16(const float* __restrict__ x, __bf16* __restrict__ xbf, int n64) {
    int i = blockIdx.x * 256 + threadIdx.x;
    if (i < n64) xbf[i] = (__bf16)x[i];
}

// ---------- 2) pack W_l / W_r into per-wave WMMA B-fragment order (bf16) ----------
// B 32x16 K order used here (16-bit, matches A loads): lanes 0-15 -> K=0..15, lanes 16-31 -> K=16..31; 16 elems/lane.
// pack index: ((mat*32 + kt*16+ct)*32 + lane)*16 + j   (j = K within half)
__global__ void pack_w(const float* __restrict__ Wl, const float* __restrict__ Wr,
                       __bf16* __restrict__ wpack) {
    int t = blockIdx.x * 256 + threadIdx.x;
    if (t >= 2048) return;
    int lane = t & 31, ct = (t >> 5) & 15, kt = (t >> 9) & 1, mat = t >> 10;
    const float* W = mat ? Wr : Wl;
    int n = ct * 16 + (lane & 15);
    int kbase = kt * 32 + ((lane < 16) ? 0 : 16);
    __bf16* dst = wpack + mat * 16384 + (((kt * 16 + ct) * 32 + lane) * 16);
#pragma unroll
    for (int j = 0; j < 16; ++j) dst[j] = (__bf16)W[(kbase + j) * HC + n];
}

// ---------- 3) GEMM: xl = x@W_l + b_l ; xr = x@W_r + b_r  (WMMA bf16 -> f32) ----------
__global__ void __launch_bounds__(128) gemm_wmma(
    const __bf16* __restrict__ xbf, const __bf16* __restrict__ wpack,
    const float* __restrict__ bl, const float* __restrict__ br,
    float* __restrict__ xl, float* __restrict__ xr, int n) {
    int wave = (blockIdx.x * blockDim.x + threadIdx.x) >> 5;
    int lane = threadIdx.x & 31;
    int mtiles = n >> 4;                 // N is a multiple of 16 (50000/16 = 3125)
    int tilesPerMat = mtiles * 16;
    if (wave >= 2 * tilesPerMat) return;
    int mat = wave / tilesPerMat;
    int t = wave - mat * tilesPerMat;
    int mt = t >> 4, nt = t & 15;

    int m = mt * 16 + (lane & 15);
    int aoff = (lane < 16) ? 0 : 16;     // A 16x32 bf16 K order used here (matches B pack): contiguous K = 16h..16h+15
    const __bf16* wp = wpack + mat * 16384;

    v8f acc = {};
#pragma unroll
    for (int kt = 0; kt < 2; ++kt) {
        const __bf16* ap = xbf + m * INC + kt * 32 + aoff;
        v8bf alo = *(const v8bf*)(ap);        // K = base..base+7
        v8bf ahi = *(const v8bf*)(ap + 8);    // K = base+8..base+15
        const __bf16* bp = wp + (((kt * 16 + nt) * 32 + lane) * 16);
        v8bf blo = *(const v8bf*)(bp);
        v8bf bhi = *(const v8bf*)(bp + 8);
        v16bf A, B;
#pragma unroll
        for (int i = 0; i < 8; ++i) { A[i] = alo[i]; A[i + 8] = ahi[i];
                                      B[i] = blo[i]; B[i + 8] = bhi[i]; }
        acc = __builtin_amdgcn_wmma_f32_16x16x32_bf16(
            /*neg_a=*/false, A, /*neg_b=*/false, B,
            /*c_mod=*/(short)0, acc, /*reuse_a=*/false, /*reuse_b=*/false);
    }
    const float* bias = mat ? br : bl;
    float* OUT = mat ? xr : xl;
    int col = nt * 16 + (lane & 15);
    float bv = bias[col];
    int rbase = mt * 16 + ((lane < 16) ? 0 : 8);   // C/D layout: VGPR v -> row v / v+8
#pragma unroll
    for (int v = 0; v < 8; ++v)
        OUT[(size_t)(rbase + v) * HC + col] = acc[v] + bv;
}

// ---------- 4) init: out accumulator = 0, encoded max = 0 (-inf), denom = 0 ----------
__global__ void init_bufs(float* __restrict__ out, unsigned* __restrict__ maxb,
                          float* __restrict__ denom, int n) {
    int i = blockIdx.x * 256 + threadIdx.x;
    int tot = n * (HC + 8);
    if (i >= tot) return;
    if (i < n * HC) { out[i] = 0.f; return; }
    i -= n * HC;
    if (i < n * 4) { maxb[i] = 0u; return; }
    denom[i - n * 4] = 0.f;
}

// ---------- 5) per-edge attention logits + segment max (one wave32 per edge) ----------
__global__ void __launch_bounds__(256) edge_logits(
    const long long* __restrict__ ei, const float* __restrict__ xl,
    const float* __restrict__ xr, const float* __restrict__ att,
    float* __restrict__ logits, unsigned* __restrict__ maxb, int E, int Et) {
    int e = (blockIdx.x * blockDim.x + threadIdx.x) >> 5;
    int lane = threadIdx.x & 31;
    if (e >= Et) return;
    long long s, d;
    if (e < E) { s = ei[e]; d = ei[E + e]; } else { s = d = (long long)(e - E); }
    int c8 = lane * 8;
    v4f a0 = *(const v4f*)(xl + s * HC + c8);
    v4f a1 = *(const v4f*)(xl + s * HC + c8 + 4);
    v4f b0 = *(const v4f*)(xr + d * HC + c8);
    v4f b1 = *(const v4f*)(xr + d * HC + c8 + 4);
    v4f t0 = *(const v4f*)(att + c8);
    v4f t1 = *(const v4f*)(att + c8 + 4);
    float sum = 0.f;
#pragma unroll
    for (int j = 0; j < 4; ++j) {
        float m0 = a0[j] + b0[j]; m0 = m0 > 0.f ? m0 : NEG_SLOPE * m0;
        float m1 = a1[j] + b1[j]; m1 = m1 > 0.f ? m1 : NEG_SLOPE * m1;
        sum += m0 * t0[j] + m1 * t1[j];
    }
    // reduce 8 lanes per head (4 heads x 8 lanes x 8 channels = 256)
    sum += __shfl_xor(sum, 1, 8);
    sum += __shfl_xor(sum, 2, 8);
    sum += __shfl_xor(sum, 4, 8);
    if ((lane & 7) == 0) {
        int h = lane >> 3;
        logits[(size_t)e * 4 + h] = sum;
        atomicMax(maxb + d * 4 + h, enc_f32(sum));
    }
}

// ---------- 6) ex = exp(logit - max), segment sum of ex ----------
__global__ void edge_exp(const long long* __restrict__ ei, float* __restrict__ logits,
                         const unsigned* __restrict__ maxb, float* __restrict__ denom,
                         int E, int Et) {
    int i = blockIdx.x * 256 + threadIdx.x;
    if (i >= Et * 4) return;
    int e = i >> 2, h = i & 3;
    long long d = (e < E) ? ei[E + e] : (long long)(e - E);
    float mx = dec_f32(maxb[d * 4 + h]);
    float ex = __expf(logits[i] - mx);
    logits[i] = ex;                         // recomputed fresh each launch -> deterministic
    atomicAdd(denom + d * 4 + h, ex);
}

// ---------- 7) scatter-aggregate: out[dst] += alpha * xl[src] (wave per edge) ----------
__global__ void __launch_bounds__(256) edge_aggr(
    const long long* __restrict__ ei, const float* __restrict__ xl,
    const float* __restrict__ logits, const float* __restrict__ denom,
    float* __restrict__ out, int E, int Et) {
    int e = (blockIdx.x * blockDim.x + threadIdx.x) >> 5;
    int lane = threadIdx.x & 31;
    if (e >= Et) return;
    long long s, d;
    if (e < E) { s = ei[e]; d = ei[E + e]; } else { s = d = (long long)(e - E); }
    int h = lane >> 3;
    float alpha = logits[(size_t)e * 4 + h] / denom[d * 4 + h];
    int c8 = lane * 8;
    v4f a0 = *(const v4f*)(xl + s * HC + c8);
    v4f a1 = *(const v4f*)(xl + s * HC + c8 + 4);
    float* op = out + d * HC + c8;
#pragma unroll
    for (int j = 0; j < 4; ++j) {
        atomicAdd(op + j,     alpha * a0[j]);
        atomicAdd(op + 4 + j, alpha * a1[j]);
    }
}

// ---------- 8) +bias, LayerNorm(256), ReLU  (wave per node, in-place in d_out) ----------
__global__ void __launch_bounds__(256) ln_relu(
    float* __restrict__ out, const float* __restrict__ bias,
    const float* __restrict__ gamma, const float* __restrict__ beta, int n) {
    int node = (blockIdx.x * blockDim.x + threadIdx.x) >> 5;
    int lane = threadIdx.x & 31;
    if (node >= n) return;
    int c8 = lane * 8;
    float* row = out + (size_t)node * HC;
    v4f v0 = *(v4f*)(row + c8);
    v4f v1 = *(v4f*)(row + c8 + 4);
    v4f bb0 = *(const v4f*)(bias + c8);
    v4f bb1 = *(const v4f*)(bias + c8 + 4);
    float s = 0.f, sq = 0.f;
#pragma unroll
    for (int j = 0; j < 4; ++j) {
        v0[j] += bb0[j]; v1[j] += bb1[j];
        s  += v0[j] + v1[j];
        sq += v0[j] * v0[j] + v1[j] * v1[j];
    }
#pragma unroll
    for (int m = 1; m < 32; m <<= 1) { s += __shfl_xor(s, m, 32); sq += __shfl_xor(sq, m, 32); }
    float mean = s * (1.f / HC);
    float var  = sq * (1.f / HC) - mean * mean;
    var = var < 0.f ? 0.f : var;
    float rstd = rsqrtf(var + LN_EPS);
    v4f g0 = *(const v4f*)(gamma + c8);
    v4f g1 = *(const v4f*)(gamma + c8 + 4);
    v4f e0 = *(const v4f*)(beta + c8);
    v4f e1 = *(const v4f*)(beta + c8 + 4);
#pragma unroll
    for (int j = 0; j < 4; ++j) {
        float y0 = (v0[j] - mean) * rstd * g0[j] + e0[j];
        float y1 = (v1[j] - mean) * rstd * g1[j] + e1[j];
        row[c8 + j]     = y0 > 0.f ? y0 : 0.f;
        row[c8 + 4 + j] = y1 > 0.f ? y1 : 0.f;
    }
}

extern "C" void kernel_launch(void* const* d_in, const int* in_sizes, int n_in,
                              void* d_out, int out_size, void* d_ws, size_t ws_size,
                              hipStream_t stream) {
    const float*     x     = (const float*)d_in[0];
    const long long* ei    = (const long long*)d_in[1];
    const float*     Wl    = (const float*)d_in[2];
    const float*     bl    = (const float*)d_in[3];
    const float*     Wr    = (const float*)d_in[4];
    const float*     br    = (const float*)d_in[5];
    const float*     att   = (const float*)d_in[6];
    const float*     bias  = (const float*)d_in[7];
    const float*     gamma = (const float*)d_in[8];
    const float*     beta  = (const float*)d_in[9];
    float* out = (float*)d_out;

    const int N  = in_sizes[0] / INC;
    const int E  = in_sizes[1] / 2;
    const int Et = E + N;

    // workspace carve-out (256B aligned)
    char* ws = (char*)d_ws;
    size_t off = 0;
    auto carve = [&](size_t bytes) -> char* {
        char* p = ws + off;
        off += (bytes + 255) & ~(size_t)255;
        return p;
    };
    __bf16*  xbf    = (__bf16*)carve((size_t)N * INC * sizeof(__bf16));
    __bf16*  wpack  = (__bf16*)carve(2 * 16384 * sizeof(__bf16));
    float*   xl     = (float*)carve((size_t)N * HC * sizeof(float));
    float*   xr     = (float*)carve((size_t)N * HC * sizeof(float));
    float*   logits = (float*)carve((size_t)Et * 4 * sizeof(float));
    unsigned* maxb  = (unsigned*)carve((size_t)N * 4 * sizeof(unsigned));
    float*   denom  = (float*)carve((size_t)N * 4 * sizeof(float));
    (void)ws_size; (void)n_in; (void)out_size;

    // 1-2) precision conversion + weight fragment packing
    int n64 = N * INC;
    cvt_x_bf16<<<(n64 + 255) / 256, 256, 0, stream>>>(x, xbf, n64);
    pack_w<<<8, 256, 0, stream>>>(Wl, Wr, wpack);

    // 3) WMMA GEMMs (both weight matrices in one grid)
    int waves   = 2 * (N / 16) * 16;
    int gthread = waves * 32;
    gemm_wmma<<<(gthread + 127) / 128, 128, 0, stream>>>(xbf, wpack, bl, br, xl, xr, N);

    // 4) zero/neg-inf init of accumulators
    int itot = N * (HC + 8);
    init_bufs<<<(itot + 255) / 256, 256, 0, stream>>>(out, maxb, denom, N);

    // 5-7) edge phase: logits+max, exp+sum, weighted scatter
    int ethreads = Et * 32;
    edge_logits<<<(ethreads + 255) / 256, 256, 0, stream>>>(ei, xl, xr, att, logits, maxb, E, Et);
    edge_exp<<<(Et * 4 + 255) / 256, 256, 0, stream>>>(ei, logits, maxb, denom, E, Et);
    edge_aggr<<<(ethreads + 255) / 256, 256, 0, stream>>>(ei, xl, logits, denom, out, E, Et);

    // 8) +bias, LayerNorm, ReLU
    int lthreads = N * 32;
    ln_relu<<<(lthreads + 255) / 256, 256, 0, stream>>>(out, bias, gamma, beta, N);
}
